// VQVAE_6150393168142
// MI455X (gfx1250) — matrix operand order repaired and checked by reading, not compiled, not run on hardware
//
#include <hip/hip_runtime.h>
#include <hip/hip_bf16.h>
#include <cstdint>

typedef __attribute__((ext_vector_type(16))) _Float16 v16h;
typedef __attribute__((ext_vector_type(8)))  _Float16 v8h;
typedef __attribute__((ext_vector_type(8)))  float    v8f;

static constexpr int B = 1024, T = 64, V = 51, E = 64, H = 256, Z = 32, D = 256;
static constexpr int QN = 8, K = 4096;
static constexpr int TM = T - 1;                  // 63 decoder steps

// d_out layout: [xloss][nonpadded][pred_tokens B*TM][Q][idx QN*B]
static constexpr int OUT_XLOSS  = 0;
static constexpr int OUT_NONPAD = 1;
static constexpr int OUT_PRED   = 2;
static constexpr int OUT_Q      = OUT_PRED + B * TM;   // 64514
static constexpr int OUT_IDX    = OUT_Q + 1;           // 64515

__device__ __forceinline__ v8f wmma16x16(v16h a, v16h b, v8f c) {
  return __builtin_amdgcn_wmma_f32_16x16x32_f16(false, a, false, b, (short)0, c,
                                                false, false);
}

// ---- f16 fragment loads (global address space, 16B vector loads) ----
typedef __attribute__((address_space(1))) const v8h* gv8p;
__device__ __forceinline__ v8h ld8(const _Float16* p) {
  return *(gv8p)(unsigned long long)p;
}
__device__ __forceinline__ v16h ld16c(const _Float16* p0, const _Float16* p1) {
  v8h lo = ld8(p0), hi = ld8(p1);
  return __builtin_shufflevector(lo, hi, 0, 1, 2, 3, 4, 5, 6, 7,
                                 8, 9, 10, 11, 12, 13, 14, 15);
}
// A-frag (16x32): lane r+16s holds row r; element e <-> K = 8s + (e<8?e:e+8)
__device__ __forceinline__ v16h frag_a16(const _Float16* base, int kb, int s) {
  return ld16c(base + kb + 8 * s, base + kb + 8 * s + 16);
}
// B-frag (32x16): lane n+16s holds col n; element e <-> K = 8s + (e<8?e:e+8)
__device__ __forceinline__ v16h frag_b16(const _Float16* base, int kb, int s) {
  return ld16c(base + kb + 8 * s, base + kb + 8 * s + 16);
}

// ---- f32-source fragments (cold kernels only) ----
template <class F>
__device__ __forceinline__ v16h frag_a32(F&& ld, int kb, int s) {
  v16h a;
#pragma unroll
  for (int e = 0; e < 8; ++e)  a[e] = (_Float16)ld(kb + 8 * s + e);
#pragma unroll
  for (int e = 8; e < 16; ++e) a[e] = (_Float16)ld(kb + 8 * s + e + 8);
  return a;
}

__device__ __forceinline__ float sigmf(float x) { return 1.f / (1.f + __expf(-x)); }

// Fused GRU step on f16 operands (f32 WMMA accumulate, f32 state update).
// Block owns 16 batch rows; 8 waves x 6 tiles cover N=768.
// r/z thirds fully fused; n third keeps input/recurrent parts separate.
template <int EX, class XR>
__device__ void gru_core(float* __restrict__ h, _Float16* __restrict__ h16,
                         const _Float16* __restrict__ Whh,
                         const _Float16* __restrict__ Wih,
                         const float* __restrict__ bih,
                         const float* __restrict__ bhh,
                         XR xrun, _Float16* __restrict__ hs16) {
  __shared__ float s_rz[16][2 * H];   // summed r,z pre-activations (32 KB)
  __shared__ float s_nx[16][H];       // n-gate input part   (16 KB)
  __shared__ float s_nh[16][H];       // n-gate recurrent part (16 KB)
  const int lane = threadIdx.x & 31;
  const int wave = threadIdx.x >> 5;
  const int s = lane >> 4;
  const int r = lane & 15;
  const int brow = blockIdx.x * 16 + r;
  const _Float16* hrow = h16 + (size_t)brow * H;

#pragma unroll
  for (int ti = 0; ti < 6; ++ti) {
    const int nt  = wave + ti * 8;        // 0..47 (wave-uniform)
    const int col = nt * 16 + r;          // gate column 0..767
    const _Float16* wh = Whh + (size_t)col * H;
    const _Float16* wi = Wih + (size_t)col * EX;
    v8f acc = {};
#pragma unroll
    for (int kb = 0; kb < H; kb += 32)
      acc = wmma16x16(frag_a16(hrow, kb, s), frag_b16(wh, kb, s), acc);
    if (nt < 32) {                        // r,z gates: fuse input GEMM
#pragma unroll
      for (int kb = 0; kb < EX; kb += 32) {
        v16h a = ld16c(xrun(kb + 8 * s), xrun(kb + 8 * s + 16));
        acc = wmma16x16(a, frag_b16(wi, kb, s), acc);
      }
      const float bsum = bih[col] + bhh[col];
#pragma unroll
      for (int i = 0; i < 8; ++i) s_rz[8 * s + i][col] = acc[i] + bsum;
    } else {                              // n gate: separate accumulators
      v8f accx = {};
#pragma unroll
      for (int kb = 0; kb < EX; kb += 32) {
        v16h a = ld16c(xrun(kb + 8 * s), xrun(kb + 8 * s + 16));
        accx = wmma16x16(a, frag_b16(wi, kb, s), accx);
      }
      const int nc = col - 2 * H;
      const float bh_ = bhh[col], bi_ = bih[col];
#pragma unroll
      for (int i = 0; i < 8; ++i) {
        s_nh[8 * s + i][nc] = acc[i]  + bh_;
        s_nx[8 * s + i][nc] = accx[i] + bi_;
      }
    }
  }
  __syncthreads();

  const int j = threadIdx.x;              // 0..255 hidden column
#pragma unroll
  for (int i = 0; i < 16; ++i) {
    const float rr = sigmf(s_rz[i][j]);
    const float zz = sigmf(s_rz[i][H + j]);
    const float nn = tanhf(s_nx[i][j] + rr * s_nh[i][j]);
    const size_t idx = (size_t)(blockIdx.x * 16 + i) * H + j;
    const float hnew = (1.f - zz) * nn + zz * h[idx];
    h[idx] = hnew;
    const _Float16 hh = (_Float16)hnew;
    h16[idx] = hh;
    if (hs16) hs16[idx] = hh;
  }
}

// Encoder step: grid.y = direction (0 fwd, 1 bwd with reversed token index).
__global__ __launch_bounds__(256) void enc_step_kernel(
    const int* __restrict__ x, const _Float16* __restrict__ emb16,
    float* __restrict__ h_f, _Float16* __restrict__ h16_f,
    const _Float16* __restrict__ Whh_f, const _Float16* __restrict__ Wih_f,
    const float* __restrict__ bih_f, const float* __restrict__ bhh_f,
    float* __restrict__ h_b, _Float16* __restrict__ h16_b,
    const _Float16* __restrict__ Whh_b, const _Float16* __restrict__ Wih_b,
    const float* __restrict__ bih_b, const float* __restrict__ bhh_b,
    int t) {
  const int lane = threadIdx.x & 31;
  const int r = lane & 15;
  const int brow = blockIdx.x * 16 + r;
  const int dir = blockIdx.y;
  const int ttok = dir ? (T - 1 - t) : t;
  const int tok = x[brow * T + ttok];
  const _Float16* er = emb16 + (size_t)tok * E;
  auto xr = [=](int k0) { return er + k0; };
  if (dir == 0)
    gru_core<E>(h_f, h16_f, Whh_f, Wih_f, bih_f, bhh_f, xr, (_Float16*)nullptr);
  else
    gru_core<E>(h_b, h16_b, Whh_b, Wih_b, bih_b, bhh_b, xr, (_Float16*)nullptr);
}

// Decoder step: dec_in = [qz | dec_emb[src_t] | lemma], writes hs16[t].
__global__ __launch_bounds__(256) void dec_step_kernel(
    const int* __restrict__ x, const _Float16* __restrict__ demb16,
    const _Float16* __restrict__ qz16, const _Float16* __restrict__ lemma16,
    float* __restrict__ h, _Float16* __restrict__ h16,
    const _Float16* __restrict__ Whh, const _Float16* __restrict__ Wih,
    const float* __restrict__ bih, const float* __restrict__ bhh,
    _Float16* __restrict__ hs16, int t) {
  const int lane = threadIdx.x & 31;
  const int r = lane & 15;
  const int brow = blockIdx.x * 16 + r;
  const int tok = x[brow * T + t];
  const _Float16* er = demb16 + (size_t)tok * E;
  const _Float16* qr = qz16 + (size_t)brow * Z;
  const _Float16* lr = lemma16 + (size_t)brow * Z;
  // 8-half runs never straddle the [0,32)[32,96)[96,128) segment boundaries
  auto xr = [=](int k0) {
    return (k0 < Z) ? qr + k0 : ((k0 < Z + E) ? er + (k0 - Z) : lr + (k0 - Z - E));
  };
  gru_core<Z + E + Z>(h, h16, Whh, Wih, bih, bhh, xr, hs16 + (size_t)t * B * D);
}

// lemma = out_h@Wlem^T + blem ; zcat = out_h@Zw^T + Zb, out_h = [hf|hb]
__global__ __launch_bounds__(256) void outh_kernel(
    const _Float16* __restrict__ hf16, const _Float16* __restrict__ hb16,
    const _Float16* __restrict__ wz16,   // [288][512] = [Wlem; Zw] f16
    const float* __restrict__ blem, const float* __restrict__ Zb,
    float* __restrict__ lemma, _Float16* __restrict__ lemma16,
    float* __restrict__ zcat) {
  constexpr int NT = (Z + QN * Z) / 16;          // 18 col tiles
  const int gt = blockIdx.x * 8 + (threadIdx.x >> 5);
  const int mt = gt / NT, nt = gt % NT;
  const int lane = threadIdx.x & 31, s = lane >> 4, r = lane & 15;
  const int brow = mt * 16 + r;
  const _Float16* hfr = hf16 + (size_t)brow * H;
  const _Float16* hbr = hb16 + (size_t)brow * H;
  const int col = nt * 16 + r;
  const _Float16* wrow = wz16 + (size_t)col * (2 * H);
  const float bias = (col < Z) ? blem[col] : Zb[col - Z];
  auto xr = [=](int k0) { return (k0 < H) ? hfr + k0 : hbr + (k0 - H); };
  v8f acc = {};
#pragma unroll
  for (int kb = 0; kb < 2 * H; kb += 32) {
    v16h a = ld16c(xr(kb + 8 * s), xr(kb + 8 * s + 16));
    acc = wmma16x16(a, frag_b16(wrow, kb, s), acc);
  }
#pragma unroll
  for (int i = 0; i < 8; ++i) {
    const int row = mt * 16 + 8 * s + i;
    const float v = acc[i] + bias;
    if (col < Z) {
      lemma[(size_t)row * Z + col] = v;
      lemma16[(size_t)row * Z + col] = (_Float16)v;
    } else {
      zcat[(size_t)row * (QN * Z) + (col - Z)] = v;
    }
  }
}

// hdec = (qz + lemma)@Wup^T + bup  (decoder initial hidden, f32 + f16 mirror)
__global__ __launch_bounds__(256) void hidden_kernel(
    const float* __restrict__ qz, const float* __restrict__ lemma,
    const _Float16* __restrict__ wup16, const float* __restrict__ bup,
    float* __restrict__ hdec, _Float16* __restrict__ hdec16) {
  const int gt = blockIdx.x * 8 + (threadIdx.x >> 5);   // 1024 tiles
  const int mt = gt >> 4, nt = gt & 15;
  const int lane = threadIdx.x & 31, s = lane >> 4, r = lane & 15;
  const int brow = mt * 16 + r;
  const float* qr = qz + (size_t)brow * Z;
  const float* lr = lemma + (size_t)brow * Z;
  const int col = nt * 16 + r;
  v16h a = frag_a32([&](int k) { return qr[k] + lr[k]; }, 0, s);
  v8f acc = {};
  acc = wmma16x16(a, frag_b16(wup16 + (size_t)col * Z, 0, s), acc);
#pragma unroll
  for (int i = 0; i < 8; ++i) {
    const size_t idx = (size_t)(mt * 16 + 8 * s + i) * D + col;
    const float v = acc[i] + bup[col];
    hdec[idx] = v;
    hdec16[idx] = (_Float16)v;
  }
}

// logits (N padded to 64 zero rows) + log-softmax + NLL + argmax per 16 rows.
__global__ __launch_bounds__(128) void logits_kernel(
    const _Float16* __restrict__ hs16, const _Float16* __restrict__ wout16,
    const float* __restrict__ bout, const int* __restrict__ x,
    float* __restrict__ out) {
  __shared__ float lg[16][64];
  const int wave = threadIdx.x >> 5;                    // 0..3 col tiles
  const int lane = threadIdx.x & 31, s = lane >> 4, r = lane & 15;
  const int m0 = blockIdx.x * 16;                       // rows m = t*B + b
  const _Float16* arow = hs16 + (size_t)(m0 + r) * D;
  const int col = wave * 16 + r;
  const _Float16* wr = wout16 + (size_t)col * D;        // zero-padded rows >= V
  v8f acc = {};
#pragma unroll
  for (int kb = 0; kb < D; kb += 32)
    acc = wmma16x16(frag_a16(arow, kb, s), frag_b16(wr, kb, s), acc);
  const float bias = (col < V) ? bout[col] : 0.f;
#pragma unroll
  for (int i = 0; i < 8; ++i) lg[8 * s + i][col] = acc[i] + bias;
  __syncthreads();

  float xls = 0.f, cnt = 0.f;
  if (threadIdx.x < 16) {
    const int i = threadIdx.x;
    const int m = m0 + i;
    const int t = m >> 10;              // m / B
    const int b = m & (B - 1);
    float mx = lg[i][0]; int pred = 0;  // first-occurrence argmax
    for (int v = 1; v < V; ++v) { const float lv = lg[i][v]; if (lv > mx) { mx = lv; pred = v; } }
    float se = 0.f;
    for (int v = 0; v < V; ++v) se += __expf(lg[i][v] - mx);
    const float lse = __logf(se) + mx;
    const int tgt = x[b * T + t + 1];
    out[OUT_PRED + b * TM + t] = (float)pred;
    if (tgt != 0) { xls = lse - lg[i][tgt]; cnt = 1.f; }
  }
  __syncthreads();
  if (threadIdx.x < 16) { lg[0][32 + threadIdx.x] = xls; lg[1][32 + threadIdx.x] = cnt; }
  __syncthreads();
  if (threadIdx.x == 0) {
    float sx = 0.f, sc = 0.f;
    for (int i = 0; i < 16; ++i) { sx += lg[0][32 + i]; sc += lg[1][32 + i]; }
    atomicAdd(&out[OUT_XLOSS], sx);
    atomicAdd(&out[OUT_NONPAD], sc);
  }
}

__global__ void cnorm_kernel(const float* __restrict__ cb, float* __restrict__ cn) {
  const int g = blockIdx.x * blockDim.x + threadIdx.x;   // QN*K
  const float* c = cb + (size_t)g * Z;
  float sum = 0.f;
#pragma unroll
  for (int zi = 0; zi < Z; ++zi) sum += c[zi] * c[zi];
  cn[g] = sum;
}

// Exact fp32 first-min VQ: score = ||c||^2 - 2 z.c (same argmin as full dist).
__global__ __launch_bounds__(256) void vq_kernel(
    const float* __restrict__ zcat, const float* __restrict__ cb,
    const float* __restrict__ cn, float* __restrict__ qz,
    float* __restrict__ out) {
  __shared__ float red[256];
  const int g = blockIdx.x * 256 + threadIdx.x;          // (q,b) flat, QN*B
  const int q = g >> 10, b = g & (B - 1);
  float zr[Z];
  const float* zp = zcat + (size_t)b * (QN * Z) + q * Z;
#pragma unroll
  for (int zi = 0; zi < Z; ++zi) zr[zi] = zp[zi];
  const float* cq  = cb + (size_t)q * K * Z;
  const float* cnq = cn + (size_t)q * K;
  float best = 3.4e38f; int bi = 0;
  for (int k = 0; k < K; ++k) {
    const float* c = cq + (size_t)k * Z;
    float dot = 0.f;
#pragma unroll
    for (int zi = 0; zi < Z; ++zi) dot += zr[zi] * c[zi];
    const float sc = cnq[k] - 2.f * dot;
    if (sc < best) { best = sc; bi = k; }
  }
  out[OUT_IDX + g] = (float)bi;
  const float* c = cq + (size_t)bi * Z;
  float l = 0.f;
#pragma unroll
  for (int zi = 0; zi < Z; ++zi) {
    const float df = c[zi] - zr[zi];
    l += df * df;
    atomicAdd(&qz[(size_t)b * Z + zi], c[zi]);           // qz = sum_q quant
  }
  red[threadIdx.x] = l * (0.3f / (float)Z);              // (0.1+0.2)*mean
  __syncthreads();
  for (int st = 128; st > 0; st >>= 1) {
    if (threadIdx.x < st) red[threadIdx.x] += red[threadIdx.x + st];
    __syncthreads();
  }
  if (threadIdx.x == 0) atomicAdd(&out[OUT_Q], red[0]);
}

__global__ void zero_kernel(float* h_f, float* h_b, _Float16* h16f,
                            _Float16* h16b, float* qz, float* out) {
  const int g = blockIdx.x * blockDim.x + threadIdx.x;
  if (g < B * H) {
    h_f[g] = 0.f; h_b[g] = 0.f;
    h16f[g] = (_Float16)0.f; h16b[g] = (_Float16)0.f;
  }
  if (g < B * Z) qz[g] = 0.f;
  if (g == 0) { out[OUT_XLOSS] = 0.f; out[OUT_NONPAD] = 0.f; out[OUT_Q] = 0.f; }
}

__global__ void cvt_kernel(const float* __restrict__ src,
                           _Float16* __restrict__ dst, int n) {
  const int g = blockIdx.x * 256 + threadIdx.x;
  if (g < n) dst[g] = (_Float16)src[g];
}

__global__ void cvt_wout_kernel(const float* __restrict__ W,
                                _Float16* __restrict__ dst) {
  const int g = blockIdx.x * 256 + threadIdx.x;   // 64*256
  const int row = g >> 8;
  dst[g] = (row < V) ? (_Float16)W[row * D + (g & 255)] : (_Float16)0.f;
}

static inline void cvt(const float* s, _Float16* d, int n, hipStream_t st) {
  cvt_kernel<<<(n + 255) / 256, 256, 0, st>>>(s, d, n);
}

extern "C" void kernel_launch(void* const* d_in, const int* in_sizes, int n_in,
                              void* d_out, int out_size, void* d_ws, size_t ws_size,
                              hipStream_t stream) {
  (void)in_sizes; (void)n_in; (void)out_size; (void)ws_size;
  const int*   x         = (const int*)d_in[0];
  const float* enc_emb   = (const float*)d_in[1];
  const float* enc_Wih_f = (const float*)d_in[2];
  const float* enc_Whh_f = (const float*)d_in[3];
  const float* enc_bih_f = (const float*)d_in[4];
  const float* enc_bhh_f = (const float*)d_in[5];
  const float* enc_Wih_b = (const float*)d_in[6];
  const float* enc_Whh_b = (const float*)d_in[7];
  const float* enc_bih_b = (const float*)d_in[8];
  const float* enc_bhh_b = (const float*)d_in[9];
  const float* Wlem      = (const float*)d_in[10];
  const float* blem      = (const float*)d_in[11];
  const float* Zw        = (const float*)d_in[12];
  const float* Zb        = (const float*)d_in[13];
  const float* codebooks = (const float*)d_in[14];
  const float* Wup       = (const float*)d_in[15];
  const float* bup       = (const float*)d_in[16];
  const float* dec_emb   = (const float*)d_in[17];
  const float* dec_Wih   = (const float*)d_in[18];
  const float* dec_Whh   = (const float*)d_in[19];
  const float* dec_bih   = (const float*)d_in[20];
  const float* dec_bhh   = (const float*)d_in[21];
  const float* Wout      = (const float*)d_in[22];
  const float* bout      = (const float*)d_in[23];
  float* out = (float*)d_out;

  // ---- workspace: f32 region then f16 region (all 16B-aligned) ----
  float* wsf   = (float*)d_ws;
  float* h_f   = wsf;  wsf += B * H;
  float* h_b   = wsf;  wsf += B * H;
  float* lemma = wsf;  wsf += B * Z;
  float* zcat  = wsf;  wsf += B * (QN * Z);
  float* qz    = wsf;  wsf += B * Z;
  float* cn    = wsf;  wsf += QN * K;
  float* hdec  = wsf;  wsf += B * D;

  _Float16* wsh     = (_Float16*)wsf;
  _Float16* h16_f   = wsh;  wsh += B * H;
  _Float16* h16_b   = wsh;  wsh += B * H;
  _Float16* hdec16  = wsh;  wsh += B * D;
  _Float16* lemma16 = wsh;  wsh += B * Z;
  _Float16* qz16    = wsh;  wsh += B * Z;
  _Float16* w_ih_f  = wsh;  wsh += 3 * H * E;
  _Float16* w_ih_b  = wsh;  wsh += 3 * H * E;
  _Float16* w_hh_f  = wsh;  wsh += 3 * H * H;
  _Float16* w_hh_b  = wsh;  wsh += 3 * H * H;
  _Float16* w_dih   = wsh;  wsh += 3 * D * (Z + E + Z);
  _Float16* w_dhh   = wsh;  wsh += 3 * D * D;
  _Float16* wz16    = wsh;  wsh += (Z + QN * Z) * (2 * H);   // [Wlem; Zw]
  _Float16* wup16   = wsh;  wsh += D * Z;
  _Float16* emb_e16 = wsh;  wsh += V * E;
  _Float16* emb_d16 = wsh;  wsh += V * E;
  _Float16* wout16  = wsh;  wsh += 64 * D;
  _Float16* hs16    = wsh;  wsh += (size_t)TM * B * D;

  zero_kernel<<<(B * H) / 256, 256, 0, stream>>>(h_f, h_b, h16_f, h16_b, qz, out);
  cnorm_kernel<<<(QN * K) / 256, 256, 0, stream>>>(codebooks, cn);

  // one-time f16 weight/embedding conversion (parallel, off the scan path)
  cvt(enc_Wih_f, w_ih_f, 3 * H * E, stream);
  cvt(enc_Wih_b, w_ih_b, 3 * H * E, stream);
  cvt(enc_Whh_f, w_hh_f, 3 * H * H, stream);
  cvt(enc_Whh_b, w_hh_b, 3 * H * H, stream);
  cvt(dec_Wih,   w_dih,  3 * D * (Z + E + Z), stream);
  cvt(dec_Whh,   w_dhh,  3 * D * D, stream);
  cvt(Wlem,      wz16,            Z * (2 * H), stream);
  cvt(Zw,        wz16 + Z * 2 * H, QN * Z * (2 * H), stream);
  cvt(Wup,       wup16,  D * Z, stream);
  cvt(enc_emb,   emb_e16, V * E, stream);
  cvt(dec_emb,   emb_d16, V * E, stream);
  cvt_wout_kernel<<<(64 * D) / 256, 256, 0, stream>>>(Wout, wout16);

  for (int t = 0; t < T; ++t)
    enc_step_kernel<<<dim3(B / 16, 2), 256, 0, stream>>>(
        x, emb_e16,
        h_f, h16_f, w_hh_f, w_ih_f, enc_bih_f, enc_bhh_f,
        h_b, h16_b, w_hh_b, w_ih_b, enc_bih_b, enc_bhh_b, t);

  outh_kernel<<<(64 * 18) / 8, 256, 0, stream>>>(h16_f, h16_b, wz16, blem, Zb,
                                                 lemma, lemma16, zcat);
  vq_kernel<<<(QN * B) / 256, 256, 0, stream>>>(zcat, codebooks, cn, qz, out);
  cvt(qz, qz16, B * Z, stream);
  hidden_kernel<<<(64 * 16) / 8, 256, 0, stream>>>(qz, lemma, wup16, bup,
                                                   hdec, hdec16);

  for (int t = 0; t < TM; ++t)
    dec_step_kernel<<<B / 16, 256, 0, stream>>>(x, emb_d16, qz16, lemma16,
                                                hdec, hdec16, w_dhh, w_dih,
                                                dec_bih, dec_bhh, hs16, t);

  logits_kernel<<<(TM * B) / 16, 128, 0, stream>>>(hs16, wout16, bout, x, out);
}
